// Model_927712935979
// MI455X (gfx1250) — matrix operand order repaired and checked by reading, not compiled, not run on hardware
//
#include <hip/hip_runtime.h>
#include <hip/hip_bf16.h>
#include <stdint.h>

// ---------------------------------------------------------------------------
// Problem constants (match the JAX reference)
// ---------------------------------------------------------------------------
constexpr int USERN = 50000;
constexpr int ITEMN = 30000;
constexpr int NNODE = USERN + ITEMN;   // 80000
constexpr int DIM   = 64;
constexpr int NEDGE = 2000000;
constexpr int IMGD  = 4096;
constexpr int TXTD  = 384;

// ---------------------------------------------------------------------------
// WMMA types (CDNA5 gfx1250, wave32)
// ---------------------------------------------------------------------------
typedef __attribute__((ext_vector_type(16))) __bf16 v16bf;
typedef __attribute__((ext_vector_type(8)))  float  v8f;

union BFrag {
    unsigned short s[16];
    unsigned int   u[8];
    v16bf          v;
};

__device__ __forceinline__ unsigned short f2bf(float f) {
    // round-to-nearest-even fp32 -> bf16 (bit-level, no __bf16 arithmetic)
    unsigned int u = __builtin_bit_cast(unsigned int, f);
    u += 0x7FFFu + ((u >> 16) & 1u);
    return (unsigned short)(u >> 16);
}

// ---------------------------------------------------------------------------
// Pack weight matrix B [K x 64] (fp32 row-major) into bf16 B-fragment layout:
// Bp[((kt*4 + nt)*32 + lane)*8 + j] = pair {B[k,n], B[k+1,n]},
//   n = nt*16 + lane%16 ; k = kt*32 + (lane/16)*16 + 2j
// so the GEMM wave reads its whole fragment as 8 consecutive dwords per lane.
// ---------------------------------------------------------------------------
__global__ void repack_b_kernel(const float* __restrict__ B,
                                unsigned int* __restrict__ Bp, int K) {
    int idx = blockIdx.x * blockDim.x + threadIdx.x;
    int total = (K >> 5) * 4 * 32 * 8;
    if (idx >= total) return;
    int j    = idx & 7;
    int lane = (idx >> 3) & 31;
    int nt   = (idx >> 8) & 3;
    int kt   = idx >> 10;
    int n = nt * 16 + (lane & 15);
    int k = kt * 32 + (lane >> 4) * 16 + 2 * j;
    unsigned int lo = f2bf(B[(size_t)k * DIM + n]);
    unsigned int hi = f2bf(B[(size_t)(k + 1) * DIM + n]);
    Bp[idx] = lo | (hi << 16);
}

// ---------------------------------------------------------------------------
// GEMM: out[M x 64] = leaky_relu(A[M x K] @ B, 0.2) using
// v_wmma_f32_16x16x32_bf16. One wave computes a 16x64 strip (4 N-tiles).
// A fragment built from 4 coalescable float4 loads per K-step.
// ---------------------------------------------------------------------------
__global__ void gemm_bf16_lrelu_kernel(const float* __restrict__ A,
                                       const unsigned int* __restrict__ Bp,
                                       float* __restrict__ out,
                                       int M, int K) {
    const int lane = threadIdx.x & 31;
    const int wv   = threadIdx.x >> 5;
    const int rt   = blockIdx.x * (blockDim.x >> 5) + wv;   // 16-row tile
    if (rt * 16 >= M) return;                               // wave-uniform exit

    const int half = lane >> 4;       // 0: lanes 0-15, 1: lanes 16-31
    const int m    = lane & 15;
    const float* Arow = A + (size_t)(rt * 16 + m) * K;

    v8f acc[4] = {v8f{0}, v8f{0}, v8f{0}, v8f{0}};
    const int KT = K >> 5;

    for (int kt = 0; kt < KT; ++kt) {
        const int kb = kt * 32 + half * 16;
        // prefetch next A chunk into cache hierarchy
        __builtin_prefetch(Arow + kb + 32, 0, 1);

        float4 a0 = *(const float4*)(Arow + kb);
        float4 a1 = *(const float4*)(Arow + kb + 4);
        float4 a2 = *(const float4*)(Arow + kb + 8);
        float4 a3 = *(const float4*)(Arow + kb + 12);

        BFrag af;
        af.s[0]  = f2bf(a0.x); af.s[1]  = f2bf(a0.y);
        af.s[2]  = f2bf(a0.z); af.s[3]  = f2bf(a0.w);
        af.s[4]  = f2bf(a1.x); af.s[5]  = f2bf(a1.y);
        af.s[6]  = f2bf(a1.z); af.s[7]  = f2bf(a1.w);
        af.s[8]  = f2bf(a2.x); af.s[9]  = f2bf(a2.y);
        af.s[10] = f2bf(a2.z); af.s[11] = f2bf(a2.w);
        af.s[12] = f2bf(a3.x); af.s[13] = f2bf(a3.y);
        af.s[14] = f2bf(a3.z); af.s[15] = f2bf(a3.w);

#pragma unroll
        for (int nt = 0; nt < 4; ++nt) {
            const unsigned int* bp = Bp + ((size_t)(kt * 4 + nt) * 32 + lane) * 8;
            BFrag bf_;
            *(uint4*)&bf_.u[0] = *(const uint4*)(bp);
            *(uint4*)&bf_.u[4] = *(const uint4*)(bp + 4);
            // 8-arg form: (neg_a, A, neg_b, B, c_mod, C, reuse_a, reuse_b)
            acc[nt] = __builtin_amdgcn_wmma_f32_16x16x32_bf16(
                false, af.v, false, bf_.v, (short)0, acc[nt], false, false);
        }
    }

    // C/D layout: VGPR r -> M = r + 8*half ; N = nt*16 + lane%16
#pragma unroll
    for (int nt = 0; nt < 4; ++nt) {
#pragma unroll
        for (int r = 0; r < 8; ++r) {
            float v = acc[nt][r];
            v = (v > 0.f) ? v : 0.2f * v;                  // LeakyReLU(0.2)
            out[(size_t)(rt * 16 + half * 8 + r) * DIM + nt * 16 + m] = v;
        }
    }
}

// ---------------------------------------------------------------------------
// Row-wise L2 normalize in place (wave per row, 2 cols/lane, wave32 shuffle)
// ---------------------------------------------------------------------------
__global__ void l2norm_rows_kernel(float* __restrict__ x, int nrows) {
    int wid  = (int)((blockIdx.x * (size_t)blockDim.x + threadIdx.x) >> 5);
    int lane = threadIdx.x & 31;
    if (wid >= nrows) return;
    float* row = x + (size_t)wid * DIM;
    float2 v = *(float2*)(row + 2 * lane);
    float s = v.x * v.x + v.y * v.y;
#pragma unroll
    for (int off = 16; off > 0; off >>= 1) s += __shfl_xor(s, off, 32);
    float inv = 1.f / fmaxf(sqrtf(s), 1e-12f);
    v.x *= inv; v.y *= inv;
    *(float2*)(row + 2 * lane) = v;
}

// ---------------------------------------------------------------------------
// SPMM: out[r] += scale * vals[e] * src[cols[e]]  (atomic scatter-add)
// src is a virtual concat: col < split -> srcLo[col], else srcHi[col-split].
// 16 lanes per edge, float4 per lane.
// ---------------------------------------------------------------------------
__global__ void spmm_kernel(const int* __restrict__ rows,
                            const int* __restrict__ cols,
                            const float* __restrict__ vals,
                            const float* __restrict__ srcLo,
                            const float* __restrict__ srcHi,
                            int split, float scale, int nEdges,
                            float* __restrict__ out) {
    long long t = (long long)blockIdx.x * blockDim.x + threadIdx.x;
    long long e = t >> 4;
    int j = (int)(t & 15);
    if (e >= nEdges) return;
    int r = rows[e];
    int c = cols[e];
    float v = vals[e] * scale;
    const float* src = (c < split) ? (srcLo + (size_t)c * DIM)
                                   : (srcHi + (size_t)(c - split) * DIM);
    float4 x = *(const float4*)(src + 4 * j);
    float* o = out + (size_t)r * DIM + 4 * j;
    atomicAdd(o + 0, v * x.x);
    atomicAdd(o + 1, v * x.y);
    atomicAdd(o + 2, v * x.z);
    atomicAdd(o + 3, v * x.w);
}

// ---------------------------------------------------------------------------
// Utility kernels
// ---------------------------------------------------------------------------
__global__ void fill0_kernel(float* __restrict__ p, int n) {
    int i = blockIdx.x * blockDim.x + threadIdx.x;
    if (i < n) p[i] = 0.f;
}

__global__ void copy_kernel(float* __restrict__ dst, const float* __restrict__ src, int n) {
    int i = blockIdx.x * blockDim.x + threadIdx.x;
    if (i < n) dst[i] = src[i];
}

// bufM = softmax(mw)[0]*bufI + softmax(mw)[1]*bufT   (elementwise)
__global__ void combine_modal_kernel(const float* __restrict__ bufI,
                                     const float* __restrict__ bufT,
                                     const float* __restrict__ mw,
                                     float* __restrict__ bufM, int n) {
    int i = blockIdx.x * blockDim.x + threadIdx.x;
    if (i >= n) return;
    float m0 = mw[0], m1 = mw[1];
    float mx = fmaxf(m0, m1);
    float e0 = __expf(m0 - mx), e1 = __expf(m1 - mx);
    float inv = 1.f / (e0 + e1);
    bufM[i] = (e0 * bufI[i] + e1 * bufT[i]) * inv;
}

// out = bufM + e1 + e2 + 0.5 * normalize(bufM)   (wave per row)
__global__ void final_out_kernel(const float* __restrict__ bufM,
                                 const float* __restrict__ e1,
                                 const float* __restrict__ e2,
                                 float* __restrict__ out, int nrows) {
    int wid  = (int)((blockIdx.x * (size_t)blockDim.x + threadIdx.x) >> 5);
    int lane = threadIdx.x & 31;
    if (wid >= nrows) return;
    size_t base = (size_t)wid * DIM + 2 * lane;
    float2 m = *(const float2*)(bufM + base);
    float s = m.x * m.x + m.y * m.y;
#pragma unroll
    for (int off = 16; off > 0; off >>= 1) s += __shfl_xor(s, off, 32);
    float inv = 0.5f / fmaxf(sqrtf(s), 1e-12f);
    float2 a = *(const float2*)(e1 + base);
    float2 b = *(const float2*)(e2 + base);
    float2 o;
    o.x = m.x + a.x + b.x + inv * m.x;
    o.y = m.y + a.y + b.y + inv * m.y;
    *(float2*)(out + base) = o;
}

// ---------------------------------------------------------------------------
// Orchestration
// ---------------------------------------------------------------------------
static inline int cdiv_ll(long long a, long long b) { return (int)((a + b - 1) / b); }

extern "C" void kernel_launch(void* const* d_in, const int* in_sizes, int n_in,
                              void* d_out, int out_size, void* d_ws, size_t ws_size,
                              hipStream_t stream) {
    const float* uEmb  = (const float*)d_in[0];
    const float* iEmb  = (const float*)d_in[1];
    const float* imgE  = (const float*)d_in[2];
    const float* txtE  = (const float*)d_in[3];
    const float* imgW  = (const float*)d_in[4];
    const float* txtW  = (const float*)d_in[5];
    const float* mw    = (const float*)d_in[6];
    const int*   aRow  = (const int*)d_in[7];
    const int*   aCol  = (const int*)d_in[8];
    const float* aVal  = (const float*)d_in[9];
    const int*   gRow  = (const int*)d_in[10];
    const int*   gCol  = (const int*)d_in[11];
    const float* gVal  = (const float*)d_in[12];
    const int*   tRow  = (const int*)d_in[13];
    const int*   tCol  = (const int*)d_in[14];
    const float* tVal  = (const float*)d_in[15];
    float*       outp  = (float*)d_out;

    // workspace carve-up
    float* ws      = (float*)d_ws;
    float* imgFeat = ws;                               // ITEM*64
    float* txtFeat = imgFeat + (size_t)ITEMN * DIM;    // ITEM*64
    float* bufI    = txtFeat + (size_t)ITEMN * DIM;    // N*64
    float* bufT    = bufI    + (size_t)NNODE * DIM;    // N*64
    float* bufM    = bufT    + (size_t)NNODE * DIM;    // N*64
    float* bufE    = bufM    + (size_t)NNODE * DIM;    // N*64 (staging + hop1)
    unsigned int* bpImg = (unsigned int*)(bufE + (size_t)NNODE * DIM); // 131072 dw
    unsigned int* bpTxt = bpImg + (IMGD / 32) * 1024;                  // 12288 dw

    const int nvec = NNODE * DIM;       // 5.12M
    const int uvec = USERN * DIM;       // 3.2M
    const int spmmBlocks = cdiv_ll((long long)NEDGE * 16, 256);

    // 1. pack weight matrices into bf16 B-fragment layout
    {
        int totI = (IMGD / 32) * 1024;
        repack_b_kernel<<<cdiv_ll(totI, 256), 256, 0, stream>>>(imgW, bpImg, IMGD);
        int totT = (TXTD / 32) * 1024;
        repack_b_kernel<<<cdiv_ll(totT, 256), 256, 0, stream>>>(txtW, bpTxt, TXTD);
    }

    // 2. dense GEMMs on WMMA + fused LeakyReLU, then row L2-normalize
    {
        int rowTiles = ITEMN / 16;                       // 1875
        int wavesPerBlock = 4;                           // blockDim 128
        int grid = cdiv_ll(rowTiles, wavesPerBlock);
        gemm_bf16_lrelu_kernel<<<grid, 128, 0, stream>>>(imgE, bpImg, imgFeat, ITEMN, IMGD);
        gemm_bf16_lrelu_kernel<<<grid, 128, 0, stream>>>(txtE, bpTxt, txtFeat, ITEMN, TXTD);
        int nwaves = cdiv_ll((long long)ITEMN * 32, 256);
        l2norm_rows_kernel<<<nwaves, 256, 0, stream>>>(imgFeat, ITEMN);
        l2norm_rows_kernel<<<nwaves, 256, 0, stream>>>(txtFeat, ITEMN);
    }

    // 3. image branch: bufI = A@[u; n(imgFeat)] + A@[prev_u; i] + 0.2 * Aimg@[u; i]
    fill0_kernel<<<cdiv_ll(nvec, 256), 256, 0, stream>>>(bufI, nvec);
    spmm_kernel<<<spmmBlocks, 256, 0, stream>>>(aRow, aCol, aVal, uEmb, imgFeat,
                                                USERN, 1.0f, NEDGE, bufI);
    copy_kernel<<<cdiv_ll(uvec, 256), 256, 0, stream>>>(bufE, bufI, uvec);
    spmm_kernel<<<spmmBlocks, 256, 0, stream>>>(aRow, aCol, aVal, bufE, iEmb,
                                                USERN, 1.0f, NEDGE, bufI);
    spmm_kernel<<<spmmBlocks, 256, 0, stream>>>(gRow, gCol, gVal, uEmb, iEmb,
                                                USERN, 0.2f, NEDGE, bufI);

    // 4. text branch: same shape into bufT
    fill0_kernel<<<cdiv_ll(nvec, 256), 256, 0, stream>>>(bufT, nvec);
    spmm_kernel<<<spmmBlocks, 256, 0, stream>>>(aRow, aCol, aVal, uEmb, txtFeat,
                                                USERN, 1.0f, NEDGE, bufT);
    copy_kernel<<<cdiv_ll(uvec, 256), 256, 0, stream>>>(bufE, bufT, uvec);
    spmm_kernel<<<spmmBlocks, 256, 0, stream>>>(aRow, aCol, aVal, bufE, iEmb,
                                                USERN, 1.0f, NEDGE, bufT);
    spmm_kernel<<<spmmBlocks, 256, 0, stream>>>(tRow, tCol, tVal, uEmb, iEmb,
                                                USERN, 0.2f, NEDGE, bufT);

    // 5. modal fusion with softmax weights
    combine_modal_kernel<<<cdiv_ll(nvec, 256), 256, 0, stream>>>(bufI, bufT, mw, bufM, nvec);

    // 6. 2-layer GCN propagation (hop1 -> bufE, hop2 -> bufI)
    fill0_kernel<<<cdiv_ll(nvec, 256), 256, 0, stream>>>(bufE, nvec);
    spmm_kernel<<<spmmBlocks, 256, 0, stream>>>(aRow, aCol, aVal, bufM, bufM,
                                                NNODE, 1.0f, NEDGE, bufE);
    fill0_kernel<<<cdiv_ll(nvec, 256), 256, 0, stream>>>(bufI, nvec);
    spmm_kernel<<<spmmBlocks, 256, 0, stream>>>(aRow, aCol, aVal, bufE, bufE,
                                                NNODE, 1.0f, NEDGE, bufI);

    // 7. out = bufM + hop1 + hop2 + 0.5 * normalize(bufM)
    {
        int nwaves = cdiv_ll((long long)NNODE * 32, 256);
        final_out_kernel<<<nwaves, 256, 0, stream>>>(bufM, bufE, bufI, outp, NNODE);
    }
}
